// GCNEncoder_74028056314069
// MI455X (gfx1250) — matrix operand order repaired and checked by reading, not compiled, not run on hardware
//
#include <hip/hip_runtime.h>

#define N_NODES 50000
#define N_EDGES 800000
#define C       128
#define C3      384
#define NTILES  (N_NODES / 16)   // 3125, exact

typedef __bf16 bf16_t;
typedef bf16_t v16bf __attribute__((ext_vector_type(16)));
typedef float  v8f   __attribute__((ext_vector_type(8)));

union BFrag { v16bf v; unsigned short u[16]; uint4 q[2]; };

__device__ __forceinline__ unsigned short f2bf(float f) {
  union { float f; unsigned int u; } x{f};
  unsigned int u = x.u;
  u += 0x7FFFu + ((u >> 16) & 1u);   // round-to-nearest-even
  return (unsigned short)(u >> 16);
}

// ---------------- weight swizzle: B matrix -> WMMA-fragment-major bf16 -------
// Output index i = ((ct*4 + ks)*32 + lane)*16 + h  maps to B[k][n] with
//   k = ks*32 + 8*(lane>=16) + (h&7) + 16*(h>=8),  n = ct*16 + (lane&15)
// trans==0: src is [K=128][N] row-major with N==128 (layer weight)
// trans==1: src is [N][K=128] row-major (GRU weight, B[k][n] = src[n*128+k])
__global__ void k_swz(const float* __restrict__ src,
                      unsigned short* __restrict__ dst, int ntiles, int trans) {
  int i = blockIdx.x * blockDim.x + threadIdx.x;
  if (i >= (ntiles << 11)) return;
  int h  = i & 15;
  int l  = (i >> 4) & 31;
  int ks = (i >> 9) & 3;
  int ct = i >> 11;
  int k  = ks * 32 + ((l >> 4) << 3) + (h & 7) + ((h >> 3) << 4);
  int n  = ct * 16 + (l & 15);
  float v = trans ? src[n * C + k] : src[k * C + n];
  dst[i] = f2bf(v);
}

__global__ void k_copy4(const float4* __restrict__ src, float4* __restrict__ dst, int n4) {
  int i = blockIdx.x * blockDim.x + threadIdx.x;
  if (i < n4) dst[i] = src[i];
}

__global__ void k_zero4(float4* __restrict__ dst, int n4) {
  int i = blockIdx.x * blockDim.x + threadIdx.x;
  if (i < n4) dst[i] = make_float4(0.f, 0.f, 0.f, 0.f);
}

// ---------------- m = x @ W  (bf16 WMMA, fp32 acc) ----------------
// block = 256 threads (8 waves); block -> one 16-row tile; wave -> one 16-col tile
__global__ __launch_bounds__(256)
void k_gemm_m(const float* __restrict__ x, const unsigned short* __restrict__ wswz,
              float* __restrict__ m) {
  __shared__ __align__(16) unsigned short sx[16 * C];   // x tile as bf16
  const int rowBase = blockIdx.x * 16;
  const int tid = threadIdx.x;
  for (int i = tid; i < 16 * C; i += 256)
    sx[i] = f2bf(x[(size_t)(rowBase + (i >> 7)) * C + (i & 127)]);
  __syncthreads();

  const int wave = tid >> 5, lane = tid & 31;
  const int aoff = (lane >> 4) << 3;             // A: +8 K for upper half-wave
  const int am = lane & 15;                      // A row / D col
  const uint4* __restrict__ wq = (const uint4*)wswz;
  v8f acc = {};
#pragma unroll
  for (int ks = 0; ks < 4; ++ks) {
    BFrag a, b;
    const uint4* ap = (const uint4*)(sx + am * C + ks * 32 + aoff);
    a.q[0] = ap[0];                              // K = ks*32+aoff .. +7
    a.q[1] = ap[2];                              // K = ks*32+aoff+16 .. +23
    const int bb = (((wave * 4 + ks) * 32) + lane) * 2;   // uint4 units
    b.q[0] = wq[bb];
    b.q[1] = wq[bb + 1];
    acc = __builtin_amdgcn_wmma_f32_16x16x32_bf16(false, a.v, false, b.v,
                                                  (short)0, acc, false, false);
  }
  const int n  = wave * 16 + am;
  const int r0 = (lane >> 4) << 3;
#pragma unroll
  for (int r = 0; r < 8; ++r)
    m[(size_t)(rowBase + r0 + r) * C + n] = acc[r];
}

// ---------------- edge gather + scatter-add ----------------
// one wave per edge; lane handles a float4 (128 floats / 32 lanes)
__global__ __launch_bounds__(256)
void k_scatter(const float* __restrict__ m, const int* __restrict__ ei,
               const float* __restrict__ ew, float* __restrict__ agg) {
  const int e = blockIdx.x * 8 + (threadIdx.x >> 5);
  const int lane = threadIdx.x & 31;
  if (e >= N_EDGES) return;
  const int   s = ei[e];
  const int   d = ei[N_EDGES + e];
  const float w = ew[e];
  const int ep = e + 4096;                      // prefetch a future source row
  if (ep < N_EDGES)
    __builtin_prefetch(m + (size_t)ei[ep] * C + lane * 4, 0, 0);
  const float4 v = ((const float4*)(m + (size_t)s * C))[lane];
  float* a = agg + (size_t)d * C + lane * 4;
  atomicAdd(a + 0, v.x * w);
  atomicAdd(a + 1, v.y * w);
  atomicAdd(a + 2, v.z * w);
  atomicAdd(a + 3, v.w * w);
}

// ---------------- fused GRU cell (2 WMMA GEMMs + gates), in-place x ----------
__global__ __launch_bounds__(256)
void k_gru(const float* __restrict__ agg, float* __restrict__ x,
           const unsigned short* __restrict__ wihS,     // swizzled, 24 tiles
           const unsigned short* __restrict__ whhS,     // swizzled, 24 tiles
           const float* __restrict__ bih, const float* __restrict__ bhh,
           int relu) {
  __shared__ __align__(16) unsigned short sA[16 * C];   // agg tile bf16
  __shared__ __align__(16) unsigned short sH[16 * C];   // x   tile bf16
  __shared__ float sGi[16 * C3];          // gi = agg@wih^T + bih
  __shared__ float sGh[16 * C3];          // gh = x  @whh^T + bhh
  const int rowBase = blockIdx.x * 16;
  const int tid = threadIdx.x;
  for (int i = tid; i < 16 * C; i += 256) {
    size_t g = (size_t)(rowBase + (i >> 7)) * C + (i & 127);
    sA[i] = f2bf(agg[g]);
    sH[i] = f2bf(x[g]);
  }
  __syncthreads();

  const int wave = tid >> 5, lane = tid & 31;
  const int aoff = (lane >> 4) << 3;
  const int am = lane & 15;
  for (int t = wave; t < 48; t += 8) {          // 24 gi tiles + 24 gh tiles
    const bool isGi = t < 24;
    const int ct = t % 24;
    const unsigned short* sa = isGi ? sA : sH;
    const uint4* __restrict__ wq = (const uint4*)(isGi ? wihS : whhS);
    v8f acc = {};
#pragma unroll
    for (int ks = 0; ks < 4; ++ks) {
      BFrag a, b;
      const uint4* ap = (const uint4*)(sa + am * C + ks * 32 + aoff);
      a.q[0] = ap[0];
      a.q[1] = ap[2];
      const int bb = (((ct * 4 + ks) * 32) + lane) * 2;
      b.q[0] = wq[bb];
      b.q[1] = wq[bb + 1];
      acc = __builtin_amdgcn_wmma_f32_16x16x32_bf16(false, a.v, false, b.v,
                                                    (short)0, acc, false, false);
    }
    const int n = ct * 16 + am;
    const float bias = isGi ? bih[n] : bhh[n];
    float* dst = isGi ? sGi : sGh;
    const int r0 = (lane >> 4) << 3;
#pragma unroll
    for (int r = 0; r < 8; ++r)
      dst[(r0 + r) * C3 + n] = acc[r] + bias;
  }
  __syncthreads();

  for (int i = tid; i < 16 * C; i += 256) {
    const int r = i >> 7, c = i & 127;
    const float ir  = sGi[r * C3 + c];
    const float iz  = sGi[r * C3 + C + c];
    const float in_ = sGi[r * C3 + 2 * C + c];
    const float hr  = sGh[r * C3 + c];
    const float hz  = sGh[r * C3 + C + c];
    const float hn  = sGh[r * C3 + 2 * C + c];
    const float rg = 1.f / (1.f + __expf(-(ir + hr)));
    const float zg = 1.f / (1.f + __expf(-(iz + hz)));
    const float ng = tanhf(in_ + rg * hn);
    const size_t g = (size_t)(rowBase + r) * C + c;
    const float h = x[g];                        // fp32 old state (own rows only)
    float out = (1.f - zg) * ng + zg * h;
    if (relu) out = fmaxf(out, 0.f);
    x[g] = out;
  }
}

// ---------------- host orchestration ----------------
extern "C" void kernel_launch(void* const* d_in, const int* in_sizes, int n_in,
                              void* d_out, int out_size, void* d_ws, size_t ws_size,
                              hipStream_t stream) {
  (void)in_sizes; (void)n_in; (void)out_size; (void)ws_size;
  const float* x0   = (const float*)d_in[0];
  const int*   ei   = (const int*)d_in[1];
  const float* ew   = (const float*)d_in[2];
  const float* w1   = (const float*)d_in[3];
  const float* wih1 = (const float*)d_in[4];
  const float* whh1 = (const float*)d_in[5];
  const float* bih1 = (const float*)d_in[6];
  const float* bhh1 = (const float*)d_in[7];
  const float* w2   = (const float*)d_in[8];
  const float* wih2 = (const float*)d_in[9];
  const float* whh2 = (const float*)d_in[10];
  const float* bih2 = (const float*)d_in[11];
  const float* bhh2 = (const float*)d_in[12];

  float* xbuf = (float*)d_out;                  // running node state (in-place)
  char* ws = (char*)d_ws;
  float* m_buf = (float*)ws;  ws += (size_t)N_NODES * C * sizeof(float);
  float* agg   = (float*)ws;  ws += (size_t)N_NODES * C * sizeof(float);
  unsigned short* w1b   = (unsigned short*)ws;  ws += (size_t)2 * C * C * 2;
  unsigned short* w2b   = (unsigned short*)ws;  ws += (size_t)2 * C * C * 2;
  unsigned short* wih1S = (unsigned short*)ws;  ws += (size_t)C * C3 * 2;
  unsigned short* whh1S = (unsigned short*)ws;  ws += (size_t)C * C3 * 2;
  unsigned short* wih2S = (unsigned short*)ws;  ws += (size_t)C * C3 * 2;
  unsigned short* whh2S = (unsigned short*)ws;  ws += (size_t)C * C3 * 2;

  // swizzle weights to WMMA-fragment-major bf16
  const int n8  = 8 << 11;    // 8 col tiles  (layer weight, per layer)
  const int n24 = 24 << 11;   // 24 col tiles (GRU weights)
  for (int layer = 0; layer < 2; ++layer) {
    k_swz<<<(n8 + 255) / 256, 256, 0, stream>>>(w1 + layer * C * C,
                                                w1b + layer * C * C, 8, 0);
    k_swz<<<(n8 + 255) / 256, 256, 0, stream>>>(w2 + layer * C * C,
                                                w2b + layer * C * C, 8, 0);
  }
  k_swz<<<(n24 + 255) / 256, 256, 0, stream>>>(wih1, wih1S, 24, 1);
  k_swz<<<(n24 + 255) / 256, 256, 0, stream>>>(whh1, whh1S, 24, 1);
  k_swz<<<(n24 + 255) / 256, 256, 0, stream>>>(wih2, wih2S, 24, 1);
  k_swz<<<(n24 + 255) / 256, 256, 0, stream>>>(whh2, whh2S, 24, 1);

  // x <- input
  const int n4 = N_NODES * C / 4;
  k_copy4<<<(n4 + 255) / 256, 256, 0, stream>>>((const float4*)x0, (float4*)xbuf, n4);

  for (int conv = 0; conv < 9; ++conv) {
    const unsigned short* wb  = (conv == 0) ? w1b   : w2b;
    const unsigned short* wiS = (conv == 0) ? wih1S : wih2S;
    const unsigned short* whS = (conv == 0) ? whh1S : whh2S;
    const float* bi = (conv == 0) ? bih1 : bih2;
    const float* bh = (conv == 0) ? bhh1 : bhh2;
    for (int layer = 0; layer < 2; ++layer) {
      k_gemm_m<<<NTILES, 256, 0, stream>>>(xbuf, wb + layer * C * C, m_buf);
      k_zero4<<<(n4 + 255) / 256, 256, 0, stream>>>((float4*)agg, n4);
      k_scatter<<<N_EDGES / 8, 256, 0, stream>>>(m_buf, ei, ew, agg);
      const int relu = (layer == 1 && conv < 8) ? 1 : 0;
      k_gru<<<NTILES, 256, 0, stream>>>(agg, xbuf, wiS, whS, bi, bh, relu);
    }
  }
}
